// PairFormerBlock_68384469286927
// MI455X (gfx1250) — compile-verified, matrix operand order checked by reading, not run on hardware
//
#include <hip/hip_runtime.h>
#include <hip/hip_bf16.h>

// ---------------------------------------------------------------------------
// PairFormer block for MI455X (gfx1250, wave32, WMMA).
// All dense linear algebra goes through one generic strided/batched WMMA GEMM
// using v_wmma_f32_16x16x32_f16 with LDS-staged f16 tiles.
// ---------------------------------------------------------------------------

typedef __attribute__((ext_vector_type(16))) _Float16 v16h;
typedef __attribute__((ext_vector_type(8)))  _Float16 v8h;
typedef __attribute__((ext_vector_type(2)))  _Float16 v2h;
typedef __attribute__((ext_vector_type(8)))  float    v8f;
typedef __attribute__((ext_vector_type(4)))  int      v4i;

#define NA 256
#define NB 256
#define D_PAIR 128
#define D_SINGLE 512
#define N_HEADS 4
#define RANK 32
#define PAIR_ELEMS (256L * 256L * 128L)
#define PAIR_ROWS  (256 * 256)

enum { EPI_NONE = 0, EPI_SIGMOID = 1, EPI_MUL = 2, EPI_SILU = 3, EPI_ADD = 4 };

// pointer casts for async global->LDS DMA builtin (expects int4* operands)
#define GPTR(p) ((v4i*)(p))
#define LPTR(p) ((__attribute__((address_space(3))) v4i*)(p))
#if __has_builtin(__builtin_amdgcn_global_load_async_to_lds_b128)
#define HAVE_ASYNC_LDS 1
#else
#define HAVE_ASYNC_LDS 0
#endif

// ---------------------------------------------------------------------------
// Generic strided, batched GEMM:  C[b] = epi( A[b] x B[b] + bias )
//   A element (m,k)  at A + b*abs_ + m*ars + k*acs
//   B element (k,n)  at B + b*bbs  + k*brs + n*bcs
//   C element (m,n)  at C + b*cbs  + m*crs + n*ccs   (Cin indexed identically)
//
// Block: 256 threads = 8 waves; block tile 64(M) x 64(N); K stepped by 32.
// A/B K-tiles staged in LDS as f16 (paired 32-bit LDS stores), cooperative
// unconditional loads (M multiple of 64, K multiple of 32 guaranteed by the
// callers; ragged N handled by clamp+select, no divergent control flow).
// Each wave builds fragments from LDS with 16B ds_load_b128 reads and issues
// 2 WMMAs/K-step. Fragment layout per CDNA5 ISA 7.12.2.
// ---------------------------------------------------------------------------
__global__ void __launch_bounds__(256) wmma_gemm_kernel(
    const float* __restrict__ A, const float* __restrict__ B,
    const float* __restrict__ bias, const float* __restrict__ Cin,
    float* __restrict__ C,
    int M, int N, int K,
    long ars, long acs, long abs_,
    long brs, long bcs, long bbs,
    long crs, long ccs, long cbs,
    int epi)
{
    // LDS staging: row stride 40 halves (80B, 16B-aligned) for bank spread.
    __shared__ __align__(16) _Float16 As[64][40];   // [m][k]
    __shared__ __align__(16) _Float16 Bs[64][40];   // [n][k]

    const int tid  = threadIdx.x;
    const int lane = tid & 31;
    const int wave = tid >> 5;
    const int mt = wave >> 2;          // 0..1 : 32-row band
    const int nt = wave & 3;           // 0..3 : 16-col band
    const int m0blk = blockIdx.y * 64;
    const int n0blk = blockIdx.x * 64;
    const long zb = blockIdx.z;

    const float* __restrict__ Ab = A + zb * abs_;
    const float* __restrict__ Bb = B + zb * bbs;

    const int l15 = lane & 15;
    const int kh  = lane >> 4;         // which K-half this lane holds

    v8f acc0 = {};
    v8f acc1 = {};

    for (int k0 = 0; k0 < K; k0 += 32) {
        // ---- stage A tile: 64x32 halves as 1024 k-pairs, 4 per thread ----
#pragma unroll
        for (int i = 0; i < 4; ++i) {
            const int p = i * 256 + tid;
            const int m = p >> 4;              // 0..63
            const int k = (p & 15) * 2;        // 0,2,..30
            const int gm = m0blk + m;
            const int mc = (gm < M) ? gm : (M - 1);
            const long abase = (long)mc * ars + (long)(k0 + k) * acs;
            float a0 = Ab[abase];
            float a1 = Ab[abase + acs];
            a0 = (gm < M) ? a0 : 0.0f;
            a1 = (gm < M) ? a1 : 0.0f;
            v2h pk; pk[0] = (_Float16)a0; pk[1] = (_Float16)a1;
            *(v2h*)&As[m][k] = pk;             // 32-bit ds_store
        }
        // ---- stage B tile: 32(k)x64(n) halves as k-pairs, 4 per thread ----
#pragma unroll
        for (int i = 0; i < 4; ++i) {
            const int p = i * 256 + tid;
            const int n = p >> 4;              // 0..63
            const int k = (p & 15) * 2;        // 0,2,..30
            const int gn = n0blk + n;
            const int nc = (gn < N) ? gn : (N - 1);
            const long bbase = (long)(k0 + k) * brs + (long)nc * bcs;
            float b0 = Bb[bbase];
            float b1 = Bb[bbase + brs];
            b0 = (gn < N) ? b0 : 0.0f;
            b1 = (gn < N) ? b1 : 0.0f;
            v2h pk; pk[0] = (_Float16)b0; pk[1] = (_Float16)b1;
            *(v2h*)&Bs[n][k] = pk;
        }
        __syncthreads();

        // ---- fragments from LDS (two b128 reads each) + 2 WMMAs ----
        // halves 0..7 hold K = kh*8 .. kh*8+7 ; halves 8..15 hold K+16.
        const int bcol = nt * 16 + l15;
        const v8h b0 = *(const v8h*)&Bs[bcol][kh * 8];
        const v8h b1 = *(const v8h*)&Bs[bcol][16 + kh * 8];
        const v16h bf = __builtin_shufflevector(b0, b1,
            0,1,2,3,4,5,6,7,8,9,10,11,12,13,14,15);

        const int arow0 = mt * 32 + l15;
        const v8h a00 = *(const v8h*)&As[arow0][kh * 8];
        const v8h a01 = *(const v8h*)&As[arow0][16 + kh * 8];
        const v16h af0 = __builtin_shufflevector(a00, a01,
            0,1,2,3,4,5,6,7,8,9,10,11,12,13,14,15);
        acc0 = __builtin_amdgcn_wmma_f32_16x16x32_f16(
            false, af0, false, bf, (short)0, acc0, false, false);

        const int arow1 = arow0 + 16;
        const v8h a10 = *(const v8h*)&As[arow1][kh * 8];
        const v8h a11 = *(const v8h*)&As[arow1][16 + kh * 8];
        const v16h af1 = __builtin_shufflevector(a10, a11,
            0,1,2,3,4,5,6,7,8,9,10,11,12,13,14,15);
        acc1 = __builtin_amdgcn_wmma_f32_16x16x32_f16(
            false, af1, false, bf, (short)0, acc1, false, false);

        __syncthreads();
    }

    // ---- epilogue: C/D layout: VGPR v -> row 8*(lane>>4)+v, col lane&15 ----
    const int n = n0blk + nt * 16 + l15;
    const bool nok = (n < N);
    const float bv = (bias && nok) ? bias[n] : 0.0f;
#pragma unroll
    for (int sub = 0; sub < 2; ++sub) {
        const v8f acc = sub ? acc1 : acc0;
#pragma unroll
        for (int v = 0; v < 8; ++v) {
            const int m = m0blk + mt * 32 + sub * 16 + kh * 8 + v;
            if (m < M && nok) {
                float val = acc[v] + bv;
                const long cidx = zb * cbs + (long)m * crs + (long)n * ccs;
                if (epi == EPI_SIGMOID)      val = 1.0f / (1.0f + __expf(-val));
                else if (epi == EPI_SILU)    val = val / (1.0f + __expf(-val));
                else if (epi == EPI_MUL)     val *= Cin[cidx];
                else if (epi == EPI_ADD)     val += Cin[cidx];
                C[cidx] = val;
            }
        }
    }
}

// ---------------------------------------------------------------------------
// LayerNorm over last dim D (D multiple of 128). One block (128 thr) per row.
// transpose=1 (pair tensors only): input row (i*256+j) written to (j*256+i).
// ---------------------------------------------------------------------------
__global__ void __launch_bounds__(128) ln_kernel(
    const float* __restrict__ x, float* __restrict__ y,
    const float* __restrict__ g, const float* __restrict__ b,
    int D, int transpose)
{
    __shared__ float red[128];
    const int row = blockIdx.x;
    long orow = row;
    if (transpose) orow = (long)(row & 255) * 256 + (row >> 8);
    const float* xr = x + (long)row * D;
    float* yr = y + orow * D;
    const int nt = D >> 7;   // elems per thread (1 for 128, 4 for 512)

    float v[4];
    float local = 0.0f;
    for (int t = 0; t < nt; ++t) { v[t] = xr[threadIdx.x + t * 128]; local += v[t]; }
    red[threadIdx.x] = local; __syncthreads();
    for (int s = 64; s > 0; s >>= 1) {
        if (threadIdx.x < s) red[threadIdx.x] += red[threadIdx.x + s];
        __syncthreads();
    }
    const float mean = red[0] / (float)D;
    __syncthreads();
    local = 0.0f;
    for (int t = 0; t < nt; ++t) { const float d = v[t] - mean; local += d * d; }
    red[threadIdx.x] = local; __syncthreads();
    for (int s = 64; s > 0; s >>= 1) {
        if (threadIdx.x < s) red[threadIdx.x] += red[threadIdx.x + s];
        __syncthreads();
    }
    const float rstd = rsqrtf(red[0] / (float)D + 1e-5f);
    for (int t = 0; t < nt; ++t) {
        const int d = threadIdx.x + t * 128;
        yr[d] = (v[t] - mean) * rstd * g[d] + b[d];
    }
}

// pair_out[r,:] = LN(pair[r,:] + upd[r',:]) ; r' transposed if flag set. D=128.
__global__ void __launch_bounds__(128) add_ln_kernel(
    const float* __restrict__ pair, const float* __restrict__ upd,
    float* __restrict__ outp,
    const float* __restrict__ g, const float* __restrict__ b, int transpose)
{
    __shared__ float red[128];
    const int row = blockIdx.x;
    long urow = row;
    if (transpose) urow = (long)(row & 255) * 256 + (row >> 8);
    const float pv = pair[(long)row * 128 + threadIdx.x];
    float v = pv + upd[urow * 128 + threadIdx.x];
    red[threadIdx.x] = v; __syncthreads();
    for (int s = 64; s > 0; s >>= 1) {
        if (threadIdx.x < s) red[threadIdx.x] += red[threadIdx.x + s];
        __syncthreads();
    }
    const float mean = red[0] / 128.0f;
    __syncthreads();
    const float d = v - mean;
    red[threadIdx.x] = d * d; __syncthreads();
    for (int s = 64; s > 0; s >>= 1) {
        if (threadIdx.x < s) red[threadIdx.x] += red[threadIdx.x + s];
        __syncthreads();
    }
    const float rstd = rsqrtf(red[0] / 128.0f + 1e-5f);
    outp[(long)row * 128 + threadIdx.x] = d * rstd * g[threadIdx.x] + b[threadIdx.x];
}

// OPM tail: pair_cur[r,:] = pair_in[r,:] + LN(tmp[r,:] + out_b) ; D=128
__global__ void __launch_bounds__(128) opm_finish_kernel(
    const float* __restrict__ tmp, const float* __restrict__ out_b,
    const float* __restrict__ g, const float* __restrict__ b,
    const float* __restrict__ pair_in, float* __restrict__ pair_out)
{
    __shared__ float red[128];
    const int row = blockIdx.x;
    float v = tmp[(long)row * 128 + threadIdx.x] + out_b[threadIdx.x];
    red[threadIdx.x] = v; __syncthreads();
    for (int s = 64; s > 0; s >>= 1) {
        if (threadIdx.x < s) red[threadIdx.x] += red[threadIdx.x + s];
        __syncthreads();
    }
    const float mean = red[0] / 128.0f;
    __syncthreads();
    const float d = v - mean;
    red[threadIdx.x] = d * d; __syncthreads();
    for (int s = 64; s > 0; s >>= 1) {
        if (threadIdx.x < s) red[threadIdx.x] += red[threadIdx.x + s];
        __syncthreads();
    }
    const float rstd = rsqrtf(red[0] / 128.0f + 1e-5f);
    const float ln = d * rstd * g[threadIdx.x] + b[threadIdx.x];
    pair_out[(long)row * 128 + threadIdx.x] =
        pair_in[(long)row * 128 + threadIdx.x] + ln;
}

__global__ void __launch_bounds__(256) ew_mul_kernel(
    float* __restrict__ a, const float* __restrict__ b, long n)
{
    const long i = (long)blockIdx.x * blockDim.x + threadIdx.x;
    if (i < n) a[i] *= b[i];
}

// ---------------------------------------------------------------------------
// Fused attention core: one block per (head h, row l). 256 threads, one query
// per thread, online softmax. K/V slice (256x32 each, f32) staged in 64KB LDS
// via async global->LDS DMA (ASYNCcnt) when the toolchain exposes it.
// Out may alias K's global buffer (compute reads only LDS).
// ---------------------------------------------------------------------------
__global__ void __launch_bounds__(256) attn_kernel(
    const float* __restrict__ Q, const float* __restrict__ Kg,
    const float* __restrict__ Vg, const float* __restrict__ Bias,
    float* __restrict__ Out)
{
    __shared__ float Ks[256 * 32];
    __shared__ float Vs[256 * 32];
    const int h = blockIdx.x;      // 0..3
    const int l = blockIdx.y;      // 0..255
    const int t = threadIdx.x;     // query index / loader row
    const long base = ((long)l * 256 + t) * 128 + h * 32;

    float q[32];
#if HAVE_ASYNC_LDS
    // async DMA: each lane copies its K/V row as 8x b128 transfers
#pragma unroll
    for (int i = 0; i < 8; ++i) {
        __builtin_amdgcn_global_load_async_to_lds_b128(
            GPTR(Kg + base + i * 4), LPTR(&Ks[t * 32 + i * 4]), 0, 0);
        __builtin_amdgcn_global_load_async_to_lds_b128(
            GPTR(Vg + base + i * 4), LPTR(&Vs[t * 32 + i * 4]), 0, 0);
    }
#pragma unroll 8
    for (int d = 0; d < 32; ++d) q[d] = Q[base + d];
#if __has_builtin(__builtin_amdgcn_s_wait_asynccnt)
    __builtin_amdgcn_s_wait_asynccnt(0);
#else
    asm volatile("s_wait_asynccnt 0x0" ::: "memory");
#endif
#else
#pragma unroll 8
    for (int d = 0; d < 32; ++d) {
        Ks[t * 32 + d] = Kg[base + d];
        Vs[t * 32 + d] = Vg[base + d];
        q[d] = Q[base + d];
    }
#endif
    __syncthreads();

    const float scale = 0.17677669529663687f;   // 1/sqrt(32)
    float mx = -1e30f, lsum = 0.0f, acc[32];
#pragma unroll 8
    for (int d = 0; d < 32; ++d) acc[d] = 0.0f;

    for (int k = 0; k < 256; ++k) {
        float s = 0.0f;
#pragma unroll
        for (int d = 0; d < 32; ++d) s += q[d] * Ks[k * 32 + d];
        s = s * scale + Bias[((long)l * 256 + k) * 4 + h];
        const float mn = fmaxf(mx, s);
        const float corr = __expf(mx - mn);
        const float p = __expf(s - mn);
        lsum = lsum * corr + p;
#pragma unroll
        for (int d = 0; d < 32; ++d) acc[d] = acc[d] * corr + p * Vs[k * 32 + d];
        mx = mn;
    }
    const float inv = 1.0f / lsum;
#pragma unroll 8
    for (int d = 0; d < 32; ++d) Out[base + d] = acc[d] * inv;
}

// ---------------------------------------------------------------------------
// Host side
// ---------------------------------------------------------------------------
static void launch_gemm(hipStream_t s, const float* A, const float* B,
                        const float* bias, const float* Cin, float* C,
                        int M, int N, int K,
                        long ars, long acs, long abs_,
                        long brs, long bcs, long bbs,
                        long crs, long ccs, long cbs,
                        int batch, int epi)
{
    dim3 g((N + 63) / 64, (M + 63) / 64, batch);
    wmma_gemm_kernel<<<g, 256, 0, s>>>(A, B, bias, Cin, C, M, N, K,
                                       ars, acs, abs_, brs, bcs, bbs,
                                       crs, ccs, cbs, epi);
}

// plain row-major GEMM, batch 1
static void gemm_rm(hipStream_t s, const float* A, const float* B,
                    const float* bias, const float* Cin, float* C,
                    int M, int N, int K, int epi)
{
    launch_gemm(s, A, B, bias, Cin, C, M, N, K,
                K, 1, 0,  N, 1, 0,  N, 1, 0,  1, epi);
}

extern "C" void kernel_launch(void* const* d_in, const int* in_sizes, int n_in,
                              void* d_out, int out_size, void* d_ws, size_t ws_size,
                              hipStream_t stream)
{
    (void)in_sizes; (void)n_in; (void)out_size; (void)ws_size;
    const float* single_a = (const float*)d_in[0];
    const float* single_b = (const float*)d_in[1];
    const float* pair_in  = (const float*)d_in[2];
#define PRM(i) ((const float*)d_in[i])

    float* out     = (float*)d_out;
    float* out_sa  = out;                 // [256,512]
    float* out_sb  = out + 131072;        // [256,512]
    float* pair    = out + 262144;        // [256,256,128] live pair accumulator

    float* w = (float*)d_ws;
    float* buf0 = w;                      // pair-sized temps
    float* buf1 = w + PAIR_ELEMS;
    float* buf2 = w + 2 * PAIR_ELEMS;
    float* buf3 = w + 3 * PAIR_ELEMS;
    float* sm   = w + 4 * PAIR_ELEMS;     // small-temp region
    float* sa_ln = sm;                    // 131072
    float* sb_ln = sm + 131072;           // 131072
    float* aproj = sm + 262144;           // 8192
    float* bproj = sm + 270336;           // 8192
    float* Mjrd  = sm + 278528;           // 1048576  [j, r, d]
    float* projB = sm + 1327104;          // 262144   [l, s, 4]
    float* t0    = sm + 1589248;          // 131072
    float* hid   = sm + 1720320;          // 524288

    // ======================= OPM =======================
    ln_kernel<<<NA, 128, 0, stream>>>(single_a, sa_ln, PRM(3), PRM(4), D_SINGLE, 0);
    ln_kernel<<<NB, 128, 0, stream>>>(single_b, sb_ln, PRM(3), PRM(4), D_SINGLE, 0);
    gemm_rm(stream, sa_ln, PRM(5), PRM(6), nullptr, aproj, NA, RANK, D_SINGLE, EPI_NONE);
    gemm_rm(stream, sb_ln, PRM(7), PRM(8), nullptr, bproj, NB, RANK, D_SINGLE, EPI_NONE);
    // M[j,r,d] = sum_s b[j,s] * out_w[r,s,d]   (batched over r)
    launch_gemm(stream, bproj, PRM(9), nullptr, nullptr, Mjrd,
                NB, D_PAIR, RANK,
                RANK, 1, 0,          // A: bproj[j,s]
                128, 1, 32L * 128,   // B: out_w[r][s,d], batch stride over r
                4096, 1, 128,        // C: Mjrd[j*4096 + r*128 + d]
                RANK, EPI_NONE);
    // out[i,j,d] = sum_r a[i,r] * M[j,r,d]     (batched over j)
    launch_gemm(stream, aproj, Mjrd, nullptr, nullptr, buf0,
                NA, D_PAIR, RANK,
                RANK, 1, 0,          // A: a[i,r]
                128, 1, 4096,        // B: M[j][r,d], batch over j
                32768, 1, 128,       // C: buf0[i*32768 + j*128 + d]
                NB, EPI_NONE);
    // pair = pair_in + LN(out + out_b)
    opm_finish_kernel<<<PAIR_ROWS, 128, 0, stream>>>(buf0, PRM(10), PRM(11), PRM(12),
                                                     pair_in, pair);

    // =================== Triangle multiplications ===================
    // +0 ng +1 nb +2 lp_w +3 lp_b +4 lg_w +5 lg_b +6 rp_w +7 rp_b
    // +8 rg_w +9 rg_b +10 cg +11 cb +12 op_w +13 op_b +14 og_w +15 og_b +16 og +17 ob
    auto tri_mul = [&](int base, bool outgoing) {
        ln_kernel<<<PAIR_ROWS, 128, 0, stream>>>(pair, buf0, PRM(base), PRM(base + 1), D_PAIR, 0);
        gemm_rm(stream, buf0, PRM(base + 4), PRM(base + 5), nullptr, buf1,
                PAIR_ROWS, D_PAIR, D_PAIR, EPI_SIGMOID);
        gemm_rm(stream, buf0, PRM(base + 2), PRM(base + 3), buf1, buf1,
                PAIR_ROWS, D_PAIR, D_PAIR, EPI_MUL);
        gemm_rm(stream, buf0, PRM(base + 8), PRM(base + 9), nullptr, buf2,
                PAIR_ROWS, D_PAIR, D_PAIR, EPI_SIGMOID);
        gemm_rm(stream, buf0, PRM(base + 6), PRM(base + 7), buf2, buf2,
                PAIR_ROWS, D_PAIR, D_PAIR, EPI_MUL);
        // tri einsum: 128 batched (d) 256x256x256 GEMMs
        if (outgoing) {
            // tri[i,j,d] = sum_k left[i,k,d]*right[j,k,d]
            launch_gemm(stream, buf1, buf2, nullptr, nullptr, buf3,
                        256, 256, 256,
                        32768, 128, 1,    // A(i,k) = left[(i*256+k)*128+d]
                        128, 32768, 1,    // B(k,j) = right[(j*256+k)*128+d]
                        32768, 128, 1,    // C(i,j) = tri[(i*256+j)*128+d]
                        D_PAIR, EPI_NONE);
        } else {
            // tri[i,j,d] = sum_k left[k,i,d]*right[k,j,d]
            launch_gemm(stream, buf1, buf2, nullptr, nullptr, buf3,
                        256, 256, 256,
                        128, 32768, 1,    // A(i,k) = left[(k*256+i)*128+d]
                        32768, 128, 1,    // B(k,j) = right[(k*256+j)*128+d]
                        32768, 128, 1,
                        D_PAIR, EPI_NONE);
        }
        ln_kernel<<<PAIR_ROWS, 128, 0, stream>>>(buf3, buf3, PRM(base + 10), PRM(base + 11), D_PAIR, 0);
        gemm_rm(stream, pair, PRM(base + 14), PRM(base + 15), nullptr, buf0,
                PAIR_ROWS, D_PAIR, D_PAIR, EPI_SIGMOID);
        gemm_rm(stream, buf3, PRM(base + 12), PRM(base + 13), buf0, buf0,
                PAIR_ROWS, D_PAIR, D_PAIR, EPI_MUL);
        add_ln_kernel<<<PAIR_ROWS, 128, 0, stream>>>(pair, buf0, pair,
                                                     PRM(base + 16), PRM(base + 17), 0);
    };
    tri_mul(13, true);    // outgoing
    tri_mul(31, false);   // incoming

    // =================== Triangle attentions ===================
    // +0 ng +1 nb +2 q_w +3 k_w +4 v_w +5 bias_w
    // +6 gate_w +7 gate_b +8 out_w +9 out_b +10 og +11 ob
    auto tri_attn = [&](int base, bool column) {
        const int tr = column ? 1 : 0;
        ln_kernel<<<PAIR_ROWS, 128, 0, stream>>>(pair, buf0, PRM(base), PRM(base + 1), D_PAIR, tr);
        gemm_rm(stream, buf0, PRM(base + 2), nullptr, nullptr, buf1, PAIR_ROWS, D_PAIR, D_PAIR, EPI_NONE);
        gemm_rm(stream, buf0, PRM(base + 3), nullptr, nullptr, buf2, PAIR_ROWS, D_PAIR, D_PAIR, EPI_NONE);
        gemm_rm(stream, buf0, PRM(base + 4), nullptr, nullptr, buf3, PAIR_ROWS, D_PAIR, D_PAIR, EPI_NONE);
        gemm_rm(stream, buf0, PRM(base + 5), nullptr, nullptr, projB, PAIR_ROWS, N_HEADS, D_PAIR, EPI_NONE);
        attn_kernel<<<dim3(N_HEADS, 256), 256, 0, stream>>>(buf1, buf2, buf3, projB, buf2);
        gemm_rm(stream, buf0, PRM(base + 6), PRM(base + 7), nullptr, buf3,
                PAIR_ROWS, D_PAIR, D_PAIR, EPI_SIGMOID);
        ew_mul_kernel<<<(PAIR_ELEMS + 255) / 256, 256, 0, stream>>>(buf2, buf3, PAIR_ELEMS);
        gemm_rm(stream, buf2, PRM(base + 8), PRM(base + 9), nullptr, buf1,
                PAIR_ROWS, D_PAIR, D_PAIR, EPI_NONE);
        add_ln_kernel<<<PAIR_ROWS, 128, 0, stream>>>(pair, buf1, pair,
                                                     PRM(base + 10), PRM(base + 11), tr);
    };
    tri_attn(49, false);  // row
    tri_attn(61, true);   // column

    // =================== Single FFNs ===================
    auto ffn = [&](const float* x, float* y, int base) {
        ln_kernel<<<256, 128, 0, stream>>>(x, t0, PRM(base), PRM(base + 1), D_SINGLE, 0);
        gemm_rm(stream, t0, PRM(base + 2), PRM(base + 3), nullptr, hid,
                256, 4 * D_SINGLE, D_SINGLE, EPI_SILU);
        gemm_rm(stream, hid, PRM(base + 4), PRM(base + 5), x, y,
                256, D_SINGLE, 4 * D_SINGLE, EPI_ADD);
    };
    ffn(single_a, out_sa, 73);
    ffn(single_b, out_sb, 79);

    // =================== Final pair LN (in place) ===================
    ln_kernel<<<PAIR_ROWS, 128, 0, stream>>>(pair, pair, PRM(85), PRM(86), D_PAIR, 0);
#undef PRM
}
